// Atom3D_72069551227068
// MI455X (gfx1250) — matrix operand order repaired and checked by reading, not compiled, not run on hardware
//
#include <hip/hip_runtime.h>
#include <stdint.h>

typedef __attribute__((ext_vector_type(16))) __bf16 v16bf;
typedef __attribute__((ext_vector_type(8)))  __bf16 v8bf;
typedef __attribute__((ext_vector_type(8)))  float  v8f;

static constexpr int NN  = 80000;
static constexpr int E1C = 120000;
static constexpr int E2C = 40000;
static constexpr int EC  = E1C + E2C;
static constexpr int GC  = 64;
static constexpr float BN_EPS = 1e-5f;

// ---------------- utility kernels ----------------

__global__ void k_zero(float* __restrict__ p, int n) {
    int i = blockIdx.x * blockDim.x + threadIdx.x;
    if (i < n) p[i] = 0.0f;
}

__global__ void k_deg(const int* __restrict__ ei_intra, const int* __restrict__ ei_inter,
                      float* __restrict__ deg) {
    int e = blockIdx.x * blockDim.x + threadIdx.x;
    if (e >= EC) return;
    int d = (e < E1C) ? ei_intra[E1C + e] : ei_inter[E2C + (e - E1C)];
    atomicAdd(&deg[d], 1.0f);
}

__global__ void k_norm(const float* __restrict__ deg, float* __restrict__ dis,
                       float* __restrict__ selfw) {
    int i = blockIdx.x * blockDim.x + threadIdx.x;
    if (i >= NN) return;
    float dv = deg[i] + 1.0f;
    dis[i]   = rsqrtf(dv);
    selfw[i] = 1.0f / dv;     // dis*dis
}

__global__ void k_ne(const int* __restrict__ ei_intra, const int* __restrict__ ei_inter,
                     const float* __restrict__ dis, float* __restrict__ ne) {
    int e = blockIdx.x * blockDim.x + threadIdx.x;
    if (e >= EC) return;
    int s, d;
    if (e < E1C) { s = ei_intra[e]; d = ei_intra[E1C + e]; }
    else         { int t = e - E1C; s = ei_inter[t]; d = ei_inter[E2C + t]; }
    ne[e] = dis[s] * dis[d];
}

__global__ void k_f32_to_bf16(const float* __restrict__ in, __bf16* __restrict__ out, int n) {
    int i = blockIdx.x * blockDim.x + threadIdx.x;
    if (i < n) out[i] = (__bf16)in[i];
}

// Wt[o][i] = (bf16) W[i][o]   (transpose + convert)
__global__ void k_wt(const float* __restrict__ W, __bf16* __restrict__ Wt, int din, int dout) {
    int idx = blockIdx.x * blockDim.x + threadIdx.x;
    if (idx >= din * dout) return;
    int i = idx % din;
    int o = idx / din;
    Wt[idx] = (__bf16)W[(size_t)i * dout + o];
}

// ---------------- WMMA GEMM with fused self-loop + bias epilogue ----------------
// hpre = A(bf16) @ W, agg = selfw[row]*hpre + bias[col]
// A: [M,K] bf16 row-major;  Wt: [Nd,K] bf16 row-major (== W transposed)
__global__ void __launch_bounds__(256)
k_gemm_wmma(const __bf16* __restrict__ A, const __bf16* __restrict__ Wt,
            const float* __restrict__ selfw, const float* __restrict__ bias,
            float* __restrict__ hpre, float* __restrict__ agg,
            int M, int K, int Nd) {
    const int lane = threadIdx.x & 31;
    const int wave = threadIdx.x >> 5;
    const int colTile = blockIdx.x;                 // Nd/16
    const int rowTile = blockIdx.y * 8 + wave;      // M/(16*8) blocks in y
    const int r    = lane & 15;
    const int half = lane >> 4;
    const int row  = rowTile * 16 + r;
    const int col  = colTile * 16 + r;

    const __bf16* arow = A  + (size_t)row * K;
    const __bf16* brow = Wt + (size_t)col * K;

    v8f acc = {};
    for (int k0 = 0; k0 < K; k0 += 32) {
        // A fragment (16x32 bf16): lanes 0-15 hold K {0..7,16..23}, lanes 16-31 hold K {8..15,24..31}
        v8bf alo = *(const v8bf*)(arow + k0 + half * 8);
        v8bf ahi = *(const v8bf*)(arow + k0 + 16 + half * 8);
        v16bf a;
#pragma unroll
        for (int i = 0; i < 8; ++i) { a[i] = alo[i]; a[8 + i] = ahi[i]; }
        // B fragment (32x16 bf16): lane = column; lanes 0-15 K {0..7,16..23}, lanes 16-31 K {8..15,24..31}
        v8bf blo = *(const v8bf*)(brow + k0 + half * 8);
        v8bf bhi = *(const v8bf*)(brow + k0 + 16 + half * 8);
        v16bf b;
#pragma unroll
        for (int i = 0; i < 8; ++i) { b[i] = blo[i]; b[8 + i] = bhi[i]; }
        acc = __builtin_amdgcn_wmma_f32_16x16x32_bf16(
            false, a, false, b, (short)0, acc, false, false);
    }

    // C/D layout: lane = N; VGPR v -> M = v (lanes 0-15) or v+8 (lanes 16-31)
    const int rbase = rowTile * 16 + half * 8;
    const float bcol = bias[col];
#pragma unroll
    for (int v = 0; v < 8; ++v) {
        int rr = rbase + v;
        size_t idx = (size_t)rr * Nd + col;
        float val = acc[v];
        hpre[idx] = val;
        agg[idx]  = selfw[rr] * val + bcol;
    }
}

// ---------------- edge scatter: agg[dst] += hpre[src] * ne ----------------
__global__ void k_scatter(const int* __restrict__ ei_intra, const int* __restrict__ ei_inter,
                          const float* __restrict__ ne, const float* __restrict__ hpre,
                          float* __restrict__ agg, int dout) {
    long long t = (long long)blockIdx.x * blockDim.x + threadIdx.x;
    long long total = (long long)EC * dout;
    if (t >= total) return;
    int e = (int)(t / dout);
    int j = (int)(t % dout);
    int s, d;
    if (e < E1C) { s = ei_intra[e]; d = ei_intra[E1C + e]; }
    else         { int u = e - E1C; s = ei_inter[u]; d = ei_inter[E2C + u]; }
    float val = hpre[(size_t)s * dout + j] * ne[e];
    atomicAdd(&agg[(size_t)d * dout + j], val);
}

// ---------------- post: activation/BN + requantize ----------------
// mode 0: relu -> bn ; mode 1: bn -> relu ; mode 2: bn only
__global__ void k_post(const float* __restrict__ agg,
                       __bf16* __restrict__ obf, float* __restrict__ of32,
                       const float* __restrict__ g, const float* __restrict__ be,
                       const float* __restrict__ m, const float* __restrict__ v,
                       int dout, int mode) {
    long long t = (long long)blockIdx.x * blockDim.x + threadIdx.x;
    long long total = (long long)NN * dout;
    if (t >= total) return;
    int j = (int)(t % dout);
    float h = agg[t];
    if (mode == 0) h = fmaxf(h, 0.0f);
    h = g[j] * (h - m[j]) * rsqrtf(v[j] + BN_EPS) + be[j];
    if (mode == 1) h = fmaxf(h, 0.0f);
    if (obf)  obf[t]  = (__bf16)h;
    if (of32) of32[t] = h;
}

// ---------------- pooling + FC ----------------
__global__ void k_pool(const int* __restrict__ batch, const float* __restrict__ h,
                       float* __restrict__ pooled) {
    long long t = (long long)blockIdx.x * blockDim.x + threadIdx.x;
    long long total = (long long)NN * 512;
    if (t >= total) return;
    int i = (int)(t / 512);
    int j = (int)(t % 512);
    atomicAdd(&pooled[(size_t)batch[i] * 512 + j], h[t]);
}

__global__ void k_fc1(const float* __restrict__ pooled, const float* __restrict__ W,
                      const float* __restrict__ b, float* __restrict__ hid) {
    int t = blockIdx.x * blockDim.x + threadIdx.x;
    if (t >= GC * 512) return;
    int g = t / 512;
    int j = t % 512;
    float s = b[j];
    const float* prow = pooled + (size_t)g * 512;
    for (int k = 0; k < 512; ++k)
        s += fmaxf(prow[k], 0.0f) * W[(size_t)k * 512 + j];
    hid[t] = fmaxf(s, 0.0f);
}

__global__ void k_fc2(const float* __restrict__ hid, const float* __restrict__ W,
                      const float* __restrict__ b, float* __restrict__ out) {
    int g = threadIdx.x;
    if (g >= GC) return;
    float s = b[0];
    const float* hrow = hid + (size_t)g * 512;
    for (int k = 0; k < 512; ++k) s += hrow[k] * W[k];
    out[g] = s;
}

// ---------------- launch ----------------

extern "C" void kernel_launch(void* const* d_in, const int* in_sizes, int n_in,
                              void* d_out, int out_size, void* d_ws, size_t ws_size,
                              hipStream_t stream) {
    (void)in_sizes; (void)n_in; (void)out_size; (void)ws_size;

    const float* x        = (const float*)d_in[0];
    const int*   ei_intra = (const int*)d_in[1];
    const int*   ei_inter = (const int*)d_in[2];
    const int*   batch    = (const int*)d_in[3];

    const float *W[5], *B[5], *Gm[5], *Be[5], *Mn[5], *Vr[5];
    for (int l = 0; l < 5; ++l) {
        int base = 4 + 6 * l;
        W[l]  = (const float*)d_in[base + 0];
        B[l]  = (const float*)d_in[base + 1];
        Gm[l] = (const float*)d_in[base + 2];
        Be[l] = (const float*)d_in[base + 3];
        Mn[l] = (const float*)d_in[base + 4];
        Vr[l] = (const float*)d_in[base + 5];
    }
    const float* fcW1 = (const float*)d_in[34];
    const float* fcb1 = (const float*)d_in[35];
    const float* fcW2 = (const float*)d_in[36];
    const float* fcb2 = (const float*)d_in[37];

    const int din[5]  = {128, 128, 256, 512, 512};
    const int dout[5] = {128, 256, 512, 512, 512};

    char* ws = (char*)d_ws;
    size_t off = 0;
    auto alloc = [&](size_t bytes) -> void* {
        void* p = ws + off;
        off += (bytes + 255) & ~(size_t)255;
        return p;
    };
    __bf16* hA   = (__bf16*)alloc((size_t)NN * 512 * sizeof(__bf16));
    float*  hpre = (float*) alloc((size_t)NN * 512 * sizeof(float));
    float*  agg  = (float*) alloc((size_t)NN * 512 * sizeof(float));
    float*  deg  = (float*) alloc((size_t)NN * sizeof(float));
    float*  dis  = (float*) alloc((size_t)NN * sizeof(float));
    float*  sw   = (float*) alloc((size_t)NN * sizeof(float));
    float*  ne   = (float*) alloc((size_t)EC * sizeof(float));
    __bf16* Wt[5];
    for (int l = 0; l < 5; ++l)
        Wt[l] = (__bf16*)alloc((size_t)din[l] * dout[l] * sizeof(__bf16));
    float* pooled = (float*)alloc((size_t)GC * 512 * sizeof(float));
    float* hid    = (float*)alloc((size_t)GC * 512 * sizeof(float));

    const int TPB = 256;
    auto blocks = [&](long long n) { return (unsigned)((n + TPB - 1) / TPB); };

    // graph norm precompute
    k_zero<<<blocks(NN), TPB, 0, stream>>>(deg, NN);
    k_deg <<<blocks(EC), TPB, 0, stream>>>(ei_intra, ei_inter, deg);
    k_norm<<<blocks(NN), TPB, 0, stream>>>(deg, dis, sw);
    k_ne  <<<blocks(EC), TPB, 0, stream>>>(ei_intra, ei_inter, dis, ne);

    // input + weights -> bf16
    k_f32_to_bf16<<<blocks((long long)NN * 128), TPB, 0, stream>>>(x, hA, NN * 128);
    for (int l = 0; l < 5; ++l)
        k_wt<<<blocks((long long)din[l] * dout[l]), TPB, 0, stream>>>(W[l], Wt[l], din[l], dout[l]);

    // 5 GCN layers
    for (int l = 0; l < 5; ++l) {
        dim3 gg(dout[l] / 16, NN / (16 * 8));   // 80000 = 128 * 625, exact
        k_gemm_wmma<<<gg, 256, 0, stream>>>(hA, Wt[l], sw, B[l], hpre, agg,
                                            NN, din[l], dout[l]);
        k_scatter<<<blocks((long long)EC * dout[l]), TPB, 0, stream>>>(
            ei_intra, ei_inter, ne, hpre, agg, dout[l]);
        int mode = (l < 3) ? 0 : (l == 3 ? 1 : 2);
        __bf16* obf  = (l < 4) ? hA : nullptr;
        float*  of32 = (l == 4) ? hpre : nullptr;
        k_post<<<blocks((long long)NN * dout[l]), TPB, 0, stream>>>(
            agg, obf, of32, Gm[l], Be[l], Mn[l], Vr[l], dout[l], mode);
    }

    // pooling + MLP head
    k_zero<<<blocks(GC * 512), TPB, 0, stream>>>(pooled, GC * 512);
    k_pool<<<blocks((long long)NN * 512), TPB, 0, stream>>>(batch, hpre, pooled);
    k_fc1 <<<blocks(GC * 512), TPB, 0, stream>>>(pooled, fcW1, fcb1, hid);
    k_fc2 <<<1, 64, 0, stream>>>(hid, fcW2, fcb2, (float*)d_out);
}
